// Transformer_33792802685820
// MI455X (gfx1250) — matrix operand order repaired and checked by reading, not compiled, not run on hardware
//
#include <hip/hip_runtime.h>
#include <hip/hip_bf16.h>

typedef __attribute__((ext_vector_type(16))) __bf16 bf16x16;
typedef __attribute__((ext_vector_type(8)))  __bf16 bf16x8;
typedef __attribute__((ext_vector_type(8)))  float  f32x8;
typedef int v4i_vs __attribute__((vector_size(16)));   // matches builtin param

#define BM 128
#define BN 128
#define BK 64      // two wmma K-steps per stage

#define GF_RELU    1
#define GF_OUTBF16 2
#define GF_TRANSC  4

#if __has_builtin(__builtin_amdgcn_global_load_async_to_lds_b128)
#define USE_ASYNC 1
#else
#define USE_ASYNC 0
#endif

#define GPTR(p) ((__attribute__((address_space(1))) v4i_vs*)(p))
#define LPTR(p) ((__attribute__((address_space(3))) v4i_vs*)(p))

__device__ __forceinline__ void wait_async_lds() {
#if USE_ASYNC
    asm volatile("s_wait_asynccnt 0x0" ::: "memory");
#endif
}

// ---------------------------------------------------------------------------
// Tiled bf16 GEMM, f32 accumulate via v_wmma_f32_16x16x32_bf16.
//   C[z] = act( alpha * A[z] @ B[z]^T + bias ),  B given as [N,K] row-major
// Block tile 128x128, 256 threads = 8 waves (wave tile 64x32 = 4x2 wmma tiles),
// double-buffered LDS staged with GLOBAL_LOAD_ASYNC_TO_LDS_B128 when available.
// ---------------------------------------------------------------------------
__global__ __launch_bounds__(256) void gemm_bf16_wmma(
    const __bf16* __restrict__ A, long long strideA,
    const __bf16* __restrict__ B, long long strideB,   // B: [N,K] row-major
    const float*  __restrict__ bias,
    void* __restrict__ Cout, long long strideC,
    int M, int N, int K, int lda, int ldb, int ldc,
    float alpha, int flags)
{
    __shared__ __align__(64) __bf16 lA[2][BM * BK];   // [m][k]
    __shared__ __align__(64) __bf16 lB[2][BN * BK];   // [n][k]

    const int t     = threadIdx.x;
    const int lane  = t & 31;          // wave32
    const int wid   = t >> 5;          // 0..7
    const int wm    = wid >> 2;        // 0..1 -> 64-row slab
    const int wn    = wid & 3;         // 0..3 -> 32-col slab
    const int mlane = lane & 15;
    const int half  = lane >> 4;

    const int nBase = blockIdx.x * BN;
    const int mBase = blockIdx.y * BM;
    const int z     = blockIdx.z;

    const __bf16* Ab = A + (long long)z * strideA;
    const __bf16* Bb = B + (long long)z * strideB;

    // Per-thread staging coordinates: 1024 16B chunks per tile, 4 per thread.
    const int rowT = t >> 3;              // 0..31 base row (stepped by 32)
    const int c8T  = (t & 7) * 8;         // k-chunk within row

    f32x8 acc[4][2];
    #pragma unroll
    for (int i = 0; i < 4; ++i)
        #pragma unroll
        for (int j = 0; j < 2; ++j)
            #pragma unroll
            for (int e = 0; e < 8; ++e) acc[i][j][e] = 0.0f;

    auto stage = [&](int k0, int buf) {
        #pragma unroll
        for (int r = 0; r < 4; ++r) {
            int row = rowT + r * 32;
#if USE_ASYNC
            __builtin_amdgcn_global_load_async_to_lds_b128(
                GPTR(&Ab[(long long)(mBase + row) * lda + k0 + c8T]),
                LPTR(&lA[buf][row * BK + c8T]), 0, 0);
            __builtin_amdgcn_global_load_async_to_lds_b128(
                GPTR(&Bb[(long long)(nBase + row) * ldb + k0 + c8T]),
                LPTR(&lB[buf][row * BK + c8T]), 0, 0);
#else
            *reinterpret_cast<bf16x8*>(&lA[buf][row * BK + c8T]) =
                *reinterpret_cast<const bf16x8*>(&Ab[(long long)(mBase + row) * lda + k0 + c8T]);
            *reinterpret_cast<bf16x8*>(&lB[buf][row * BK + c8T]) =
                *reinterpret_cast<const bf16x8*>(&Bb[(long long)(nBase + row) * ldb + k0 + c8T]);
#endif
        }
    };

    const int nk = K / BK;
    stage(0, 0);

    for (int i = 0; i < nk; ++i) {
        const int buf = i & 1;
        wait_async_lds();
        __syncthreads();
        if (i + 1 < nk) stage((i + 1) * BK, buf ^ 1);

        #pragma unroll
        for (int ks = 0; ks < BK; ks += 32) {
            // A fragments: ISA 16-bit A 16x32 layout
            // lane half=0: K = 0..7 then 16..23 ; half=1: K = 8..15 then 24..31
            bf16x16 afrag[4];
            #pragma unroll
            for (int mt = 0; mt < 4; ++mt) {
                int row = wm * 64 + mt * 16 + mlane;
                bf16x8 lo = *reinterpret_cast<const bf16x8*>(&lA[buf][row * BK + ks + half * 8]);
                bf16x8 hi = *reinterpret_cast<const bf16x8*>(&lA[buf][row * BK + ks + 16 + half * 8]);
                #pragma unroll
                for (int e = 0; e < 8; ++e) { afrag[mt][e] = lo[e]; afrag[mt][8 + e] = hi[e]; }
            }
            // B fragments: same K order as A (half=0: K = 0..7 then 16..23 ; half=1: K = 8..15 then 24..31)
            bf16x16 bfrag[2];
            #pragma unroll
            for (int nt = 0; nt < 2; ++nt) {
                int col = wn * 32 + nt * 16 + mlane;
                bf16x8 blo = *reinterpret_cast<const bf16x8*>(&lB[buf][col * BK + ks + half * 8]);
                bf16x8 bhi = *reinterpret_cast<const bf16x8*>(&lB[buf][col * BK + ks + 16 + half * 8]);
                #pragma unroll
                for (int e = 0; e < 8; ++e) { bfrag[nt][e] = blo[e]; bfrag[nt][8 + e] = bhi[e]; }
            }
            #pragma unroll
            for (int mt = 0; mt < 4; ++mt)
                #pragma unroll
                for (int nt = 0; nt < 2; ++nt)
                    acc[mt][nt] = __builtin_amdgcn_wmma_f32_16x16x32_bf16(
                        false, afrag[mt], false, bfrag[nt], (short)0, acc[mt][nt],
                        false, false);
        }
    }

    // ---- epilogue: C layout VGPR i -> row = i + 8*(lane>=16), col = lane&15 ----
    const bool relu   = (flags & GF_RELU)    != 0;
    const bool ob16   = (flags & GF_OUTBF16) != 0;
    const bool transC = (flags & GF_TRANSC)  != 0;
    float*  cf = reinterpret_cast<float*>(Cout)  + (long long)z * strideC;
    __bf16* ch = reinterpret_cast<__bf16*>(Cout) + (long long)z * strideC;
    #pragma unroll
    for (int mt = 0; mt < 4; ++mt) {
        #pragma unroll
        for (int nt = 0; nt < 2; ++nt) {
            int col = nBase + wn * 32 + nt * 16 + mlane;
            float bval = bias ? bias[col] : 0.0f;
            #pragma unroll
            for (int i = 0; i < 8; ++i) {
                int row = mBase + wm * 64 + mt * 16 + i + half * 8;
                float v = acc[mt][nt][i] * alpha + bval;
                if (relu) v = v > 0.0f ? v : 0.0f;
                long long o = transC ? ((long long)col * ldc + row)
                                     : ((long long)row * ldc + col);
                if (ob16) ch[o] = (__bf16)v; else cf[o] = v;
            }
        }
    }
}

// ---------------------------------------------------------------------------
// h[b,s,:] = bf16( emb[x[b,s],:] + pos_enc[s,:] )
// ---------------------------------------------------------------------------
__global__ void embed_kernel(const long long* __restrict__ x,
                             const float* __restrict__ emb,
                             const float* __restrict__ pos,
                             __bf16* __restrict__ hbf,
                             int S, int D)
{
    int bs = blockIdx.x;            // b*S + s
    int s  = bs % S;
    long long tok = x[bs];
    const float* e = emb + tok * (long long)D;
    const float* p = pos + (long long)s * D;
    __bf16* o = hbf + (long long)bs * D;
    for (int d = threadIdx.x; d < D; d += blockDim.x)
        o[d] = (__bf16)(e[d] + p[d]);
}

// ---------------------------------------------------------------------------
// f32 [R,C] -> bf16 [C,R] transpose+convert (batched over blockIdx.z)
// ---------------------------------------------------------------------------
__global__ __launch_bounds__(256) void cvt_t_kernel(
    const float* __restrict__ in, __bf16* __restrict__ out, int R, int C)
{
    __shared__ float tile[32][33];
    const float* ib = in  + (long long)blockIdx.z * R * C;
    __bf16*      ob = out + (long long)blockIdx.z * R * C;
    int r0 = blockIdx.y * 32, c0 = blockIdx.x * 32;
    int tx = threadIdx.x & 31, ty = threadIdx.x >> 5;   // 8 rows of 32
    #pragma unroll
    for (int i = ty; i < 32; i += 8)
        tile[i][tx] = ib[(long long)(r0 + i) * C + c0 + tx];
    __syncthreads();
    #pragma unroll
    for (int i = ty; i < 32; i += 8)
        ob[(long long)(c0 + i) * R + r0 + tx] = (__bf16)tile[tx][i];
}

// ---------------------------------------------------------------------------
// Masked softmax over one row (b,i): keep j >= i (reference keeps triu k=0),
// write bf16 attention weights.
// ---------------------------------------------------------------------------
__global__ __launch_bounds__(256) void softmax_mask_kernel(
    const float* __restrict__ scores, __bf16* __restrict__ attn, int S)
{
    int bi = blockIdx.x;            // b*S + i
    int i  = bi % S;
    const float* row  = scores + (long long)bi * S;
    __bf16*      orow = attn   + (long long)bi * S;
    __shared__ float red[256];
    int t = threadIdx.x;

    float m = -__builtin_inff();
    for (int j = t; j < S; j += 256)
        if (j >= i) m = fmaxf(m, row[j]);
    red[t] = m; __syncthreads();
    for (int o = 128; o > 0; o >>= 1) {
        if (t < o) red[t] = fmaxf(red[t], red[t + o]);
        __syncthreads();
    }
    m = red[0]; __syncthreads();

    float s = 0.0f;
    for (int j = t; j < S; j += 256)
        if (j >= i) s += __expf(row[j] - m);
    red[t] = s; __syncthreads();
    for (int o = 128; o > 0; o >>= 1) {
        if (t < o) red[t] += red[t + o];
        __syncthreads();
    }
    float inv = 1.0f / red[0];

    for (int j = t; j < S; j += 256)
        orow[j] = (j >= i) ? (__bf16)(__expf(row[j] - m) * inv) : (__bf16)0.0f;
}

// ---------------------------------------------------------------------------
extern "C" void kernel_launch(void* const* d_in, const int* in_sizes, int n_in,
                              void* d_out, int out_size, void* d_ws, size_t ws_size,
                              hipStream_t stream) {
    (void)in_sizes; (void)n_in; (void)out_size; (void)ws_size;
    const int D = 1024, L = 4, MF = 4096, S = 2048, Bn = 4;
    const int BS = Bn * S;                  // 8192 rows

    const long long* x = (const long long*)d_in[0];
    const float* emb = (const float*)d_in[1];
    const float* pos = (const float*)d_in[2];
    const float* wq  = (const float*)d_in[3];
    const float* bq  = (const float*)d_in[4];
    const float* wk  = (const float*)d_in[5];
    const float* bk  = (const float*)d_in[6];
    const float* wv  = (const float*)d_in[7];
    const float* bv  = (const float*)d_in[8];
    const float* w1s = (const float*)d_in[9];
    const float* b1s = (const float*)d_in[10];
    const float* w2s = (const float*)d_in[11];
    const float* b2s = (const float*)d_in[12];

    // ---- workspace layout (256B aligned) ----
    char* ws = (char*)d_ws;
    size_t off = 0;
    auto alloc = [&](size_t bytes) -> char* {
        char* p = ws + off;
        off = (off + bytes + 255) & ~(size_t)255;
        return p;
    };
    __bf16* hbf    = (__bf16*)alloc((size_t)BS * D * 2);        // embed out
    __bf16* qbf    = (__bf16*)alloc((size_t)BS * D * 2);        // [B*S, D]
    __bf16* kbf    = (__bf16*)alloc((size_t)BS * D * 2);        // [B*S, D]
    __bf16* vTbf   = (__bf16*)alloc((size_t)BS * D * 2);        // per-batch [D, S]
    float*  scores = (float*) alloc((size_t)Bn * S * S * 4);
    __bf16* attnbf = (__bf16*)alloc((size_t)Bn * S * S * 2);
    __bf16* t0bf   = (__bf16*)alloc((size_t)BS * MF * 2);       // FFN mid
    __bf16* h2bf   = (__bf16*)alloc((size_t)BS * D * 2);        // ping
    __bf16* h3bf   = (__bf16*)alloc((size_t)BS * D * 2);        // pong
    __bf16* wqT    = (__bf16*)alloc((size_t)D * D * 2);         // [D,D] (N=K)
    __bf16* wkT    = (__bf16*)alloc((size_t)D * D * 2);
    __bf16* wvT    = (__bf16*)alloc((size_t)D * D * 2);
    __bf16* w1T    = (__bf16*)alloc((size_t)L * D * MF * 2);    // per layer [MF, D]
    __bf16* w2T    = (__bf16*)alloc((size_t)L * MF * D * 2);    // per layer [D, MF]

    // ---- weight convert + transpose to [N,K] bf16 ----
    {
        dim3 gdd(D / 32, D / 32, 1);
        cvt_t_kernel<<<gdd, 256, 0, stream>>>(wq, wqT, D, D);
        cvt_t_kernel<<<gdd, 256, 0, stream>>>(wk, wkT, D, D);
        cvt_t_kernel<<<gdd, 256, 0, stream>>>(wv, wvT, D, D);
        dim3 g1(MF / 32, D / 32, L);                // [D,MF] -> [MF,D]
        cvt_t_kernel<<<g1, 256, 0, stream>>>(w1s, w1T, D, MF);
        dim3 g2(D / 32, MF / 32, L);                // [MF,D] -> [D,MF]
        cvt_t_kernel<<<g2, 256, 0, stream>>>(w2s, w2T, MF, D);
    }

    // ---- embedding + positional encoding ----
    embed_kernel<<<BS, 256, 0, stream>>>(x, emb, pos, hbf, S, D);

    // ---- Q/K projections: [8192,1024] @ [1024,1024]^T(N,K) + bias -> bf16 ----
    {
        dim3 g(D / BN, BS / BM, 1);
        gemm_bf16_wmma<<<g, 256, 0, stream>>>(hbf, 0, wqT, 0, bq, qbf, 0,
            BS, D, D, D, D, D, 1.0f, GF_OUTBF16);
        gemm_bf16_wmma<<<g, 256, 0, stream>>>(hbf, 0, wkT, 0, bk, kbf, 0,
            BS, D, D, D, D, D, 1.0f, GF_OUTBF16);
    }
    // ---- V projection, output transposed per batch: vT[b] = (h[b] @ wv)^T  [D,S]
    {
        dim3 g(D / BN, S / BM, Bn);
        gemm_bf16_wmma<<<g, 256, 0, stream>>>(
            hbf, (long long)S * D, wvT, 0, bv, vTbf, (long long)D * S,
            S, D, D, D, D, /*ldcT=*/S, 1.0f, GF_OUTBF16 | GF_TRANSC);
    }

    // ---- scores = (q @ k^T) / sqrt(D), batched over B (k is [N,K] already) ----
    {
        dim3 g(S / BN, S / BM, Bn);
        gemm_bf16_wmma<<<g, 256, 0, stream>>>(
            qbf, (long long)S * D, kbf, (long long)S * D, nullptr,
            scores, (long long)S * S,
            S, S, D, D, D, S, 0.03125f, 0);         // 1/sqrt(1024)
    }

    // ---- masked softmax (keep j >= i) -> bf16 attn ----
    softmax_mask_kernel<<<Bn * S, 256, 0, stream>>>(scores, attnbf, S);

    // ---- h = relu(attn @ v): B operand = vT [D,S] per batch ----
    {
        dim3 g(D / BN, S / BM, Bn);
        gemm_bf16_wmma<<<g, 256, 0, stream>>>(
            attnbf, (long long)S * S, vTbf, (long long)D * S, nullptr,
            h2bf, (long long)S * D,
            S, D, S, /*lda=*/S, /*ldb=*/S, /*ldc=*/D, 1.0f, GF_RELU | GF_OUTBF16);
    }

    // ---- 4 FFN layers; final write is f32 -> d_out ----
    __bf16* cur = h2bf;
    __bf16* nxt = h3bf;
    for (int l = 0; l < L; ++l) {
        dim3 g1(MF / BN, BS / BM, 1);
        gemm_bf16_wmma<<<g1, 256, 0, stream>>>(
            cur, 0, w1T + (long long)l * MF * D, 0, b1s + (long long)l * MF,
            t0bf, 0, BS, MF, D, D, /*ldb=*/D, MF, 1.0f, GF_RELU | GF_OUTBF16);

        dim3 g2(D / BN, BS / BM, 1);
        if (l < L - 1) {
            gemm_bf16_wmma<<<g2, 256, 0, stream>>>(
                t0bf, 0, w2T + (long long)l * D * MF, 0, b2s + (long long)l * D,
                nxt, 0, BS, D, MF, MF, /*ldb=*/MF, D, 1.0f, GF_RELU | GF_OUTBF16);
            __bf16* tmp = cur; cur = nxt; nxt = tmp;
        } else {
            gemm_bf16_wmma<<<g2, 256, 0, stream>>>(
                t0bf, 0, w2T + (long long)l * D * MF, 0, b2s + (long long)l * D,
                d_out, 0, BS, D, MF, MF, /*ldb=*/MF, D, 1.0f, GF_RELU);
        }
    }
}
